// TitansLinearAttention_89258010346166
// MI455X (gfx1250) — compile-verified, matrix operand order checked by reading, not run on hardware
//
#include <hip/hip_runtime.h>
#include <hip/hip_bf16.h>
#include <math.h>

// Problem constants (match reference)
static constexpr int Bb = 4, Tt = 2048, Cc = 1024, Hh = 16, Dd = 64, CS = 16;
static constexpr int NT = Tt / CS;                 // 128 chunks
static constexpr int MROWS = Bb * Tt;              // 8192
static constexpr float EPSc = 1e-5f;

typedef __attribute__((ext_vector_type(16))) __bf16 v16bf;
typedef __attribute__((ext_vector_type(8)))  float  v8f;
typedef __attribute__((ext_vector_type(4)))  unsigned v4u;
typedef __attribute__((ext_vector_type(8)))  int   v8i;
typedef __attribute__((ext_vector_type(4)))  int   v4i;

#if defined(__has_builtin)
#if __has_builtin(__builtin_amdgcn_tensor_load_to_lds)
#define HAVE_TDM 1
#endif
#endif
#ifndef HAVE_TDM
#define HAVE_TDM 0
#endif

// ---------------------------------------------------------------------------
// Generic GEMM: C[M,N] = A[M,K] @ W[K,N] + bias[N]
// fp32 in/out, bf16 WMMA compute, fp32 accumulate.
// Block tile 128(M) x 128(N), K-step 32, 256 threads (8 waves).
// Wave grid 4(M) x 2(N); each wave computes 32x64 via 2 A-frags x 4 B-frags
// = 8 v_wmma_f32_16x16x32_bf16 per K-step.
// Double-buffered LDS with register staging: global loads for tile i+1 are
// issued before the WMMAs of tile i, conversion+LDS stores happen after.
// Requires: M % 128 == 0, K % 32 == 0 (true for all launches here).
// ---------------------------------------------------------------------------
#define GTM 128
#define GTN 128
#define GTK 32

__device__ __forceinline__ void gemm_stage_load(const float* __restrict__ A,
                                                const float* __restrict__ W,
                                                int blockM, int blockN,
                                                int N, int K, int k0, bool fullN,
                                                int tid, float4 ra[4], float4 rb[4])
{
    #pragma unroll
    for (int i = 0; i < 4; ++i) {
        int idx = tid + i * 256;              // 0..1023
        int m   = idx >> 3;                   // 0..127
        int kk  = (idx & 7) * 4;              // 0..28
        ra[i] = *(const float4*)(A + (size_t)(blockM + m) * K + (k0 + kk));
    }
    if (fullN) {
        #pragma unroll
        for (int i = 0; i < 4; ++i) {
            int idx = tid + i * 256;
            int kk  = idx >> 5;               // 0..31
            int n   = (idx & 31) * 4;         // 0..124
            rb[i] = *(const float4*)(W + (size_t)(k0 + kk) * N + (blockN + n));
        }
    } else {
        #pragma unroll
        for (int i = 0; i < 4; ++i) {
            int idx = tid + i * 256;
            int kk  = idx >> 5;
            int n   = (idx & 31) * 4;
            const float* src = W + (size_t)(k0 + kk) * N + (blockN + n);
            rb[i].x = (blockN + n + 0 < N) ? src[0] : 0.f;
            rb[i].y = (blockN + n + 1 < N) ? src[1] : 0.f;
            rb[i].z = (blockN + n + 2 < N) ? src[2] : 0.f;
            rb[i].w = (blockN + n + 3 < N) ? src[3] : 0.f;
        }
    }
}

__device__ __forceinline__ void gemm_stage_store(__bf16 (*As)[GTK], __bf16 (*Bs)[GTK],
                                                 int tid, const float4 ra[4],
                                                 const float4 rb[4])
{
    #pragma unroll
    for (int i = 0; i < 4; ++i) {
        int idx = tid + i * 256;
        int m   = idx >> 3;
        int kk  = (idx & 7) * 4;
        As[m][kk + 0] = (__bf16)ra[i].x;
        As[m][kk + 1] = (__bf16)ra[i].y;
        As[m][kk + 2] = (__bf16)ra[i].z;
        As[m][kk + 3] = (__bf16)ra[i].w;
    }
    #pragma unroll
    for (int i = 0; i < 4; ++i) {
        int idx = tid + i * 256;
        int kk  = idx >> 5;
        int n   = (idx & 31) * 4;
        Bs[n + 0][kk] = (__bf16)rb[i].x;
        Bs[n + 1][kk] = (__bf16)rb[i].y;
        Bs[n + 2][kk] = (__bf16)rb[i].z;
        Bs[n + 3][kk] = (__bf16)rb[i].w;
    }
}

__global__ __launch_bounds__(256)
void titans_gemm_bf16_wmma(const float* __restrict__ A,
                           const float* __restrict__ W,
                           const float* __restrict__ bias,
                           float* __restrict__ C,
                           int M, int N, int K)
{
    __shared__ __bf16 As[2][GTM][GTK];
    __shared__ __bf16 Bs[2][GTN][GTK];   // transposed: Bs[buf][n][k] = W[k][n]

    const int tid  = threadIdx.x;
    const int lane = tid & 31;
    const int wid  = tid >> 5;
    const int waveM = wid >> 1;          // 0..3
    const int waveN = wid & 1;           // 0..1
    const int hi   = lane >> 4;          // lane half
    const int l15  = lane & 15;

    const int blockM = blockIdx.y * GTM;
    const int blockN = blockIdx.x * GTN;
    const bool fullN = (blockN + GTN) <= N;   // uniform across block

    v8f acc[2][4];
    #pragma unroll
    for (int i = 0; i < 2; ++i)
        #pragma unroll
        for (int j = 0; j < 4; ++j)
            acc[i][j] = v8f{};

    float4 ra[4], rb[4];

    // prologue: stage tile 0
    gemm_stage_load(A, W, blockM, blockN, N, K, 0, fullN, tid, ra, rb);
    gemm_stage_store(As[0], Bs[0], tid, ra, rb);
    __syncthreads();

    const int KITER = K / GTK;
    for (int it = 0; it < KITER; ++it) {
        const int cur = it & 1;
        const int nxt = cur ^ 1;
        const bool has_next = (it + 1) < KITER;

        // issue global loads for the next tile (latency hidden by WMMAs)
        if (has_next)
            gemm_stage_load(A, W, blockM, blockN, N, K, (it + 1) * GTK, fullN,
                            tid, ra, rb);

        // fragments per ISA 16-bit layout:
        // lane -> row/col = l15; element e -> K = e + (e>=8 ? 8 : 0) + hi*8
        v16bf afrag[2], bfrag[4];
        #pragma unroll
        for (int e = 0; e < 16; ++e) {
            int kk = e + ((e >= 8) ? 8 : 0) + hi * 8;
            afrag[0][e] = As[cur][waveM * 32 +      l15][kk];
            afrag[1][e] = As[cur][waveM * 32 + 16 + l15][kk];
            bfrag[0][e] = Bs[cur][waveN * 64 +      l15][kk];
            bfrag[1][e] = Bs[cur][waveN * 64 + 16 + l15][kk];
            bfrag[2][e] = Bs[cur][waveN * 64 + 32 + l15][kk];
            bfrag[3][e] = Bs[cur][waveN * 64 + 48 + l15][kk];
        }
        #pragma unroll
        for (int i = 0; i < 2; ++i)
            #pragma unroll
            for (int j = 0; j < 4; ++j)
                acc[i][j] = __builtin_amdgcn_wmma_f32_16x16x32_bf16(
                    false, afrag[i], false, bfrag[j], (short)0, acc[i][j], false, false);

        if (has_next)
            gemm_stage_store(As[nxt], Bs[nxt], tid, ra, rb);
        __syncthreads();
    }

    // ---- store: C/D layout: VGPR r -> M = r + 8*hi, N = lane&15 ----
    #pragma unroll
    for (int i = 0; i < 2; ++i) {
        int rowbase = blockM + waveM * 32 + i * 16 + hi * 8;
        #pragma unroll
        for (int j = 0; j < 4; ++j) {
            int col = blockN + waveN * 64 + j * 16 + l15;
            if (col < N) {
                float bv = bias[col];
                #pragma unroll
                for (int r = 0; r < 8; ++r)
                    C[(size_t)(rowbase + r) * N + col] = acc[i][j][r] + bv;
            }
        }
    }
}

// ---------------------------------------------------------------------------
// TDM helper: issue a tensor_load_to_lds of a 2D tile (16 rows x 64 fp32,
// row stride = 3*Cc floats) from `gaddr` into LDS offset `lds_off`.
// Descriptor per CDNA5 D# spec (groups 2/3 zero-filled; 2D tensor).
// ---------------------------------------------------------------------------
#if HAVE_TDM
__device__ __forceinline__ void tdm_load_tile_16x64(unsigned long long gaddr,
                                                    unsigned lds_off)
{
    v4u g0;
    g0[0] = 1u;                                            // count=1, user mode
    g0[1] = lds_off;                                       // lds_addr
    g0[2] = (unsigned)(gaddr & 0xFFFFFFFFu);               // global_addr[31:0]
    g0[3] = (unsigned)((gaddr >> 32) & 0x01FFFFFFu)        // global_addr[56:32]
          | (2u << 30);                                    // type = 2 (image)

    v8i g1;
    g1[0] = (int)(2u << 16);                               // data_size = 4 bytes
    g1[1] = (int)((64u & 0xFFFFu) << 16);                  // tensor_dim0 = 64 (lo16)
    g1[2] = (int)((16u & 0xFFFFu) << 16);                  // dim0 hi=0 | tensor_dim1 = 16
    g1[3] = (int)(64u << 16);                              // dim1 hi=0 | tile_dim0 = 64
    g1[4] = (int)(16u);                                    // tile_dim1 = 16, tile_dim2 = 0
    g1[5] = (int)(3 * Cc);                                 // tensor_dim0_stride = 3072
    g1[6] = 0;                                             // stride hi | dim1_stride lo
    g1[7] = 0;

    v4i gz = {0, 0, 0, 0};
#if __clang_major__ >= 23
    v8i gz8 = {0, 0, 0, 0, 0, 0, 0, 0};
    __builtin_amdgcn_tensor_load_to_lds(g0, g1, gz, gz, gz8, 0);
#else
    __builtin_amdgcn_tensor_load_to_lds(g0, g1, gz, gz, 0);
#endif
}
#endif

// ---------------------------------------------------------------------------
// Titans chunked recurrent scan. One block per (b,h); 256 threads.
// Thread tid: column j = tid>>2 (0..63), row-block rb = tid&3 (rows rb*16..+15).
// S, m held in registers. q,k chunk tiles staged to LDS by the Tensor Data
// Mover (tensor_load_to_lds, TENSORcnt) when available, else by async copies;
// v tile via async global->LDS copies (global_load_async_to_lds_b128).
// ---------------------------------------------------------------------------
__global__ __launch_bounds__(256)
void titans_scan_kernel(const float* __restrict__ qkv,   // [B,T,3C]
                        const float* __restrict__ pbuf,  // [B,T,H*3]
                        const float* __restrict__ lnw,   // [H,D]
                        const float* __restrict__ lnb,   // [H,D]
                        float* __restrict__ obuf)        // [B,T,C]
{
    const int bh = blockIdx.x;
    const int b  = bh >> 4;
    const int h  = bh & 15;
    const int tid = threadIdx.x;
    const int j   = tid >> 2;     // column of S (0..63)
    const int rb  = tid & 3;      // row block (rows rb*16 .. rb*16+15)

    float Sreg[16], mreg[16];
    #pragma unroll
    for (int r = 0; r < 16; ++r) { Sreg[r] = 0.f; mreg[r] = 0.f; }

    __shared__ float qs[16][64];
    __shared__ float ks[16][64];
    __shared__ float vs[16][64];
    __shared__ float kh[16][64];
    __shared__ float dkh[16][64];
    __shared__ float osc[16][64];
    __shared__ float pchunk[16][3];
    __shared__ float wrow[64], brow[64];

    if (tid < 64) {
        wrow[tid] = lnw[h * 64 + tid];
        brow[tid] = lnb[h * 64 + tid];
    }

    // per-thread async-copy slot: token ct, 4 floats at d4
    const int ct = tid >> 4;            // 0..15
    const int d4 = (tid & 15) * 4;      // 0..60
    const unsigned ldv   = (unsigned)(uintptr_t)&vs[ct][d4];
#if HAVE_TDM
    const unsigned ldsq0 = (unsigned)(uintptr_t)&qs[0][0];
    const unsigned ldsk0 = (unsigned)(uintptr_t)&ks[0][0];
#else
    const unsigned ldq   = (unsigned)(uintptr_t)&qs[ct][d4];
    const unsigned ldk   = (unsigned)(uintptr_t)&ks[ct][d4];
#endif

    __syncthreads();

    for (int c = 0; c < NT; ++c) {
        const int t0 = c * CS;
        const size_t rowbase = (size_t)(b * Tt + t0) * (3 * Cc) + h * 64;
        const size_t base    = rowbase + (size_t)ct * (3 * Cc) + d4;

#if HAVE_TDM
        // ---- TDM: q and k tiles (wave 0 only; EXEC-independent DMA) ----
        if (tid < 32) {
            tdm_load_tile_16x64((unsigned long long)(qkv + rowbase), ldsq0);
            tdm_load_tile_16x64((unsigned long long)(qkv + rowbase + Cc), ldsk0);
        }
#else
        // ---- fallback: async global -> LDS copies for q and k ----
        {
            unsigned long long gq = (unsigned long long)(qkv + base);
            unsigned long long gk = (unsigned long long)(qkv + base + Cc);
            asm volatile("global_load_async_to_lds_b128 %0, %1, off"
                         :: "v"(ldq), "v"(gq) : "memory");
            asm volatile("global_load_async_to_lds_b128 %0, %1, off"
                         :: "v"(ldk), "v"(gk) : "memory");
        }
#endif
        // ---- async global -> LDS copy of the v tile ----
        {
            unsigned long long gv = (unsigned long long)(qkv + base + 2 * Cc);
            asm volatile("global_load_async_to_lds_b128 %0, %1, off"
                         :: "v"(ldv), "v"(gv) : "memory");
        }
        // ---- gate params (tiny): normal loads + sigmoid ----
        if (tid < 48) {
            int t = tid / 3, cp = tid % 3;
            float v = pbuf[(size_t)(b * Tt + t0 + t) * (Hh * 3) + h * 3 + cp];
            pchunk[t][cp] = 1.f / (1.f + expf(-v));
        }
#if HAVE_TDM
        __builtin_amdgcn_s_wait_tensorcnt(0);
#endif
        asm volatile("s_wait_asynccnt 0x0" ::: "memory");
        __syncthreads();

        // ---- L2-normalize q and k rows ----
        if (tid < 32) {
            int row = tid & 15;
            float* ptr = (tid >> 4) ? ks[row] : qs[row];
            float ss = 0.f;
            for (int d = 0; d < 64; ++d) ss += ptr[d] * ptr[d];
            float scale = 1.f / fmaxf(sqrtf(ss), 1e-12f);
            for (int d = 0; d < 64; ++d) ptr[d] *= scale;
        }
        __syncthreads();

        // ---- kh = k_chunk (16x64) @ S (64x64), S at chunk start ----
        #pragma unroll 4
        for (int t = 0; t < 16; ++t) {
            float p = 0.f;
            #pragma unroll
            for (int r = 0; r < 16; ++r) p += ks[t][rb * 16 + r] * Sreg[r];
            p += __shfl_xor(p, 1);
            p += __shfl_xor(p, 2);
            if (rb == 0) kh[t][j] = p;
        }
        __syncthreads();

        // ---- LN forward + backward per row: dkh ----
        if (tid < 16) {
            int t = tid;
            float mu = 0.f;
            for (int d = 0; d < 64; ++d) mu += kh[t][d];
            mu *= (1.f / 64.f);
            float var = 0.f;
            for (int d = 0; d < 64; ++d) { float xc = kh[t][d] - mu; var += xc * xc; }
            var *= (1.f / 64.f);
            float rstd = rsqrtf(var + EPSc);
            float c1 = 0.f, c2 = 0.f;
            for (int d = 0; d < 64; ++d) {
                float xhat = (kh[t][d] - mu) * rstd;
                float dy   = 2.f * ((xhat * wrow[d] + brow[d]) - vs[t][d]);
                float wdy  = dy * wrow[d];
                c1 += xhat * wdy;
                c2 += wdy;
            }
            c1 *= (1.f / 64.f);
            c2 *= (1.f / 64.f);
            for (int d = 0; d < 64; ++d) {
                float xhat = (kh[t][d] - mu) * rstd;
                float wdy  = 2.f * ((xhat * wrow[d] + brow[d]) - vs[t][d]) * wrow[d];
                dkh[t][d]  = (wdy - xhat * c1 - c2) * rstd;
            }
        }
        __syncthreads();

        // ---- 16-token scan: m = et*m - th*g ; S = (1-al)*S + m ; o_t = q_t @ S_t
        #pragma unroll 1
        for (int t = 0; t < 16; ++t) {
            float th = pchunk[t][0];
            float al = pchunk[t][1];
            float et = pchunk[t][2];
            float dk = dkh[t][j];
            float po = 0.f;
            #pragma unroll
            for (int r = 0; r < 16; ++r) {
                float g = ks[t][rb * 16 + r] * dk;          // outer(k, dkh)
                mreg[r] = et * mreg[r] - th * g;
                Sreg[r] = (1.f - al) * Sreg[r] + mreg[r];
                po += qs[t][rb * 16 + r] * Sreg[r];
            }
            po += __shfl_xor(po, 1);
            po += __shfl_xor(po, 2);
            if (rb == 0) osc[t][j] = po;
        }
        __syncthreads();

        // ---- final LN over D, write o into [B,T,C] layout (float4 stores) ----
        if (tid < 16) {
            int t = tid;
            float mu = 0.f;
            for (int d = 0; d < 64; ++d) mu += osc[t][d];
            mu *= (1.f / 64.f);
            float var = 0.f;
            for (int d = 0; d < 64; ++d) { float xc = osc[t][d] - mu; var += xc * xc; }
            var *= (1.f / 64.f);
            float rstd = rsqrtf(var + EPSc);
            float* dst = obuf + (size_t)(b * Tt + t0 + t) * Cc + h * 64;
            for (int d = 0; d < 64; d += 4) {
                float4 o4;
                o4.x = (osc[t][d + 0] - mu) * rstd * wrow[d + 0] + brow[d + 0];
                o4.y = (osc[t][d + 1] - mu) * rstd * wrow[d + 1] + brow[d + 1];
                o4.z = (osc[t][d + 2] - mu) * rstd * wrow[d + 2] + brow[d + 2];
                o4.w = (osc[t][d + 3] - mu) * rstd * wrow[d + 3] + brow[d + 3];
                *(float4*)(dst + d) = o4;
            }
        }
        __syncthreads();   // before LDS reuse next chunk
    }
}

// ---------------------------------------------------------------------------
extern "C" void kernel_launch(void* const* d_in, const int* in_sizes, int n_in,
                              void* d_out, int out_size, void* d_ws, size_t ws_size,
                              hipStream_t stream)
{
    const float* x       = (const float*)d_in[0];
    const float* W_attn  = (const float*)d_in[1];
    const float* b_attn  = (const float*)d_in[2];
    const float* W_param = (const float*)d_in[3];
    const float* b_param = (const float*)d_in[4];
    const float* W_proj  = (const float*)d_in[5];
    const float* b_proj  = (const float*)d_in[6];
    const float* w_ln    = (const float*)d_in[7];
    const float* b_ln    = (const float*)d_in[8];
    float* out = (float*)d_out;

    float* qkvb = (float*)d_ws;                                  // [8192, 3072]
    float* pb   = qkvb + (size_t)MROWS * (3 * Cc);               // [8192, 48]
    float* ob   = pb   + (size_t)MROWS * (Hh * 3);               // [8192, 1024]

    dim3 blk(256);

    // qkv = x @ W_attn + b_attn        (8192 x 3072 x 1024)
    titans_gemm_bf16_wmma<<<dim3((3 * Cc) / GTN, MROWS / GTM), blk, 0, stream>>>(
        x, W_attn, b_attn, qkvb, MROWS, 3 * Cc, Cc);

    // p = x @ W_param + b_param        (8192 x 48 x 1024), N padded to 128
    titans_gemm_bf16_wmma<<<dim3(1, MROWS / GTM), blk, 0, stream>>>(
        x, W_param, b_param, pb, MROWS, Hh * 3, Cc);

    // chunked Titans recurrent scan    (one block per (b,h))
    titans_scan_kernel<<<dim3(Bb * Hh), blk, 0, stream>>>(qkvb, pb, w_ln, b_ln, ob);

    // out = o @ W_proj + b_proj        (8192 x 1024 x 1024)
    titans_gemm_bf16_wmma<<<dim3(Cc / GTN, MROWS / GTM), blk, 0, stream>>>(
        ob, W_proj, b_proj, out, MROWS, Cc, Cc);
}
